// PrototypicalNetwork_61847529062978
// MI455X (gfx1250) — matrix operand order repaired and checked by reading, not compiled, not run on hardware
//
#include <hip/hip_runtime.h>

#define BATCH 16
#define NSUP 4096
#define NQ   4096
#define DIM  512
#define NCLS 128
#define TILE_Q 64
#define PITCH_H 40              // LDS row pitch in halves (80 B: 16B-aligned, 20-bank stride)

typedef __attribute__((ext_vector_type(16))) __bf16 v16bf;
typedef __attribute__((ext_vector_type(8)))  float  v8f;

__device__ __forceinline__ unsigned short f2bf(float x) {
    union { float f; unsigned int u; } v; v.f = x;
    unsigned int u = v.u;
    u += 0x7FFFu + ((u >> 16) & 1u);   // round to nearest even
    return (unsigned short)(u >> 16);
}

// async 16B/lane global -> LDS copy (ASYNCcnt-tracked), GVS addressing
#define ASYNC_B128(ldsoff, voff, sbase)                                  \
    asm volatile("global_load_async_to_lds_b128 %0, %1, %2"              \
                 :: "v"(ldsoff), "v"(voff), "s"(sbase) : "memory")

// ---------------- 1) zero prototype sums + counts ----------------
__global__ __launch_bounds__(256) void pn_zero(float* __restrict__ p, int n) {
    int i = blockIdx.x * 256 + threadIdx.x;
    if (i < n) p[i] = 0.0f;
}

// ---------------- 2) scatter-add prototypes ----------------
__global__ __launch_bounds__(256) void pn_scatter(const float* __restrict__ sup,
                                                  const int* __restrict__ stgt,
                                                  float* __restrict__ sums,
                                                  float* __restrict__ counts) {
    int blk = blockIdx.x;
    int b = blk >> 12;
    int n = blk & (NSUP - 1);
    int tgt = stgt[(size_t)b * NSUP + n];
    const float* src = sup + ((size_t)b * NSUP + n) * DIM;
    float* dst = sums + ((size_t)b * NCLS + tgt) * DIM;
    int tid = threadIdx.x;
    atomicAdd(&dst[tid],       src[tid]);
    atomicAdd(&dst[tid + 256], src[tid + 256]);
    if (tid == 0) atomicAdd(&counts[b * NCLS + tgt], 1.0f);
}

// ---------------- 3) finalize: proto = sum/max(cnt,1) -> bf16, + ||p||^2 ----------------
__global__ __launch_bounds__(256) void pn_finalize(const float* __restrict__ sums,
                                                   const float* __restrict__ counts,
                                                   unsigned short* __restrict__ protoBf,
                                                   float* __restrict__ p_sq) {
    int bc = blockIdx.x;
    int tid = threadIdx.x;
    float inv = 1.0f / fmaxf(counts[bc], 1.0f);
    const float* s = sums + (size_t)bc * DIM;
    unsigned short* o = protoBf + (size_t)bc * DIM;
    float p0 = s[tid] * inv, p1 = s[tid + 256] * inv;
    o[tid] = f2bf(p0); o[tid + 256] = f2bf(p1);
    __shared__ float red[256];
    red[tid] = p0 * p0 + p1 * p1;
    __syncthreads();
    for (int st = 128; st > 0; st >>= 1) {
        if (tid < st) red[tid] += red[tid + st];
        __syncthreads();
    }
    if (tid == 0) p_sq[bc] = red[0];
}

// ---------------- 4) ||q||^2 + bf16 query copy ----------------
__global__ __launch_bounds__(256) void pn_qprep(const float* __restrict__ qe,
                                                float* __restrict__ q_sq,
                                                unsigned short* __restrict__ qBf) {
    int bq = blockIdx.x;
    int tid = threadIdx.x;
    const float* s = qe + (size_t)bq * DIM;
    unsigned short* o = qBf + (size_t)bq * DIM;
    float a = s[tid], b = s[tid + 256];
    o[tid] = f2bf(a); o[tid + 256] = f2bf(b);
    __shared__ float red[256];
    red[tid] = a * a + b * b;
    __syncthreads();
    for (int st = 128; st > 0; st >>= 1) {
        if (tid < st) red[tid] += red[tid + st];
        __syncthreads();
    }
    if (tid == 0) q_sq[bq] = red[0];
}

// ---------------- 5) fused WMMA distance + argmin + log-softmax ----------------
// block = (batch b, 64-query tile); 8 waves; double-buffered async LDS staging.
__global__ __launch_bounds__(256) void pn_gemm(const unsigned short* __restrict__ qBf,
                                               const int* __restrict__ q_tgt,
                                               const unsigned short* __restrict__ protoBf,
                                               const float* __restrict__ p_sq,
                                               const float* __restrict__ q_sq,
                                               float* __restrict__ preds_out,
                                               float* __restrict__ loss_part,
                                               float* __restrict__ corr_part) {
    // layout: bufA 2x(128x40)h @0..20480 | bufQ 2x(64x40)h @20480..30720
    //         distF (128x66)f aliases @0..33792 | red @33792..34304
    __shared__ __align__(16) unsigned char smem[34304];
    unsigned short* bufA = (unsigned short*)smem;
    unsigned short* bufQ = (unsigned short*)(smem + 20480);
    float* distF = (float*)smem;
    float* redL  = (float*)(smem + 33792);
    float* redC  = (float*)(smem + 33792 + 256);

    const int tid  = threadIdx.x;
    const int wave = tid >> 5;
    const int lane = tid & 31;
    const int lr   = lane & 15;
    const int hi   = lane >> 4;

    const int b  = blockIdx.x / (NQ / TILE_Q);
    const int q0 = (blockIdx.x % (NQ / TILE_Q)) * TILE_Q;

    // --- async staging geometry: 16B (8 halves) per lane, 4 lanes per 32-half row ---
    const int chunk = lane & 3;                 // which 8-half segment of the row
    const int rA0   = (wave * 2 + 0) * 8 + (lane >> 2);   // proto rows, instr 0
    const int rA1   = (wave * 2 + 1) * 8 + (lane >> 2);   // proto rows, instr 1
    const int rQ    = wave * 8 + (lane >> 2);             // query rows
    const unsigned long long pBase =
        (unsigned long long)(protoBf + (size_t)b * NCLS * DIM);
    const unsigned long long qBase =
        (unsigned long long)(qBf + ((size_t)b * NQ + q0) * DIM);
    const unsigned voffA0 = (unsigned)(rA0 * DIM + 8 * chunk) * 2u;
    const unsigned voffA1 = (unsigned)(rA1 * DIM + 8 * chunk) * 2u;
    const unsigned voffQ  = (unsigned)(rQ  * DIM + 8 * chunk) * 2u;
    unsigned ldsA0[2], ldsA1[2], ldsQ[2];
#pragma unroll
    for (int u = 0; u < 2; ++u) {
        ldsA0[u] = (unsigned)(size_t)(bufA + u * (NCLS * PITCH_H)   + rA0 * PITCH_H + 8 * chunk);
        ldsA1[u] = (unsigned)(size_t)(bufA + u * (NCLS * PITCH_H)   + rA1 * PITCH_H + 8 * chunk);
        ldsQ[u]  = (unsigned)(size_t)(bufQ + u * (TILE_Q * PITCH_H) + rQ  * PITCH_H + 8 * chunk);
    }
    auto issue = [&](int buf, int kstep) {
        unsigned kb = (unsigned)kstep * 64u;    // 32 halves = 64 bytes per k-step
        ASYNC_B128(ldsA0[buf], voffA0 + kb, pBase);
        ASYNC_B128(ldsA1[buf], voffA1 + kb, pBase);
        ASYNC_B128(ldsQ[buf],  voffQ  + kb, qBase);
    };

    union AB { v16bf v; unsigned int u[8]; };
    union CD { v8f  v; float f[8]; };
    CD acc[4];
#pragma unroll
    for (int t = 0; t < 4; ++t)
#pragma unroll
        for (int r = 0; r < 8; ++r) acc[t].f[r] = 0.0f;

    issue(0, 0);                                // prologue prefetch
    const int NIT = DIM / 32;                   // 16
    for (int it = 0; it < NIT; ++it) {
        const int cur = it & 1;
        if (it + 1 < NIT) {
            issue(1 - cur, it + 1);
            asm volatile("s_wait_asynccnt 0x3" ::: "memory");  // current buffer done
        } else {
            asm volatile("s_wait_asynccnt 0x0" ::: "memory");
        }
        __syncthreads();

        // A fragment: 16x32 bf16, M = wave*16 + lr
        AB a;
        {
            const unsigned short* pr =
                bufA + cur * (NCLS * PITCH_H) + (wave * 16 + lr) * PITCH_H;
            const int kb = hi ? 8 : 0;
#pragma unroll
            for (int j = 0; j < 4; ++j)
                a.u[j] = *(const unsigned int*)&pr[kb + 2 * j];
#pragma unroll
            for (int j = 0; j < 4; ++j)
                a.u[4 + j] = *(const unsigned int*)&pr[16 + kb + 2 * j];
        }
#pragma unroll
        for (int t = 0; t < 4; ++t) {
            AB bm;
            const unsigned short* qr =
                bufQ + cur * (TILE_Q * PITCH_H) + (t * 16 + lr) * PITCH_H;
            const int kb = hi ? 8 : 0;
#pragma unroll
            for (int j = 0; j < 8; ++j)
                bm.u[j] = *(const unsigned int*)&qr[kb + 2 * (j & 3) + (j >> 2) * 16];
            acc[t].v = __builtin_amdgcn_wmma_f32_16x16x32_bf16(
                false, a.v, false, bm.v, (short)0, acc[t].v, false, false);
        }
        __syncthreads();                        // buffer may be refilled next iter
    }

    // epilogue: dist = ||p||^2 + ||q||^2 - 2 p.q -> LDS (aliases staging buffers)
#pragma unroll
    for (int t = 0; t < 4; ++t) {
        float qs = q_sq[(size_t)b * NQ + q0 + t * 16 + lr];
#pragma unroll
        for (int r = 0; r < 8; ++r) {
            int c = wave * 16 + r + hi * 8;     // C/D layout: VGPR r, halves split at M=8
            float dv = p_sq[b * NCLS + c] + qs - 2.0f * acc[t].f[r];
            distF[c * 66 + t * 16 + lr] = dv;
        }
    }
    __syncthreads();

    if (tid < TILE_Q) {
        int q = q0 + tid;
        float dmin = distF[tid];
        int amin = 0;
        for (int c = 1; c < NCLS; ++c) {
            float dv = distF[c * 66 + tid];
            if (dv < dmin) { dmin = dv; amin = c; }
        }
        float s = 0.0f;
        for (int c = 0; c < NCLS; ++c)
            s += __expf(dmin - distF[c * 66 + tid]);
        int tgt = q_tgt[(size_t)b * NQ + q];
        float nll = (distF[tgt * 66 + tid] - dmin) + __logf(s);
        preds_out[(size_t)b * NQ + q] = (float)amin;
        redL[tid] = nll;
        redC[tid] = (amin == tgt) ? 1.0f : 0.0f;
    }
    __syncthreads();
    for (int st = 32; st > 0; st >>= 1) {
        if (tid < st) { redL[tid] += redL[tid + st]; redC[tid] += redC[tid + st]; }
        __syncthreads();
    }
    if (tid == 0) {
        loss_part[blockIdx.x] = redL[0];
        corr_part[blockIdx.x] = redC[0];
    }
}

// ---------------- 6) deterministic final reduce ----------------
__global__ __launch_bounds__(256) void pn_reduce(const float* __restrict__ lp,
                                                 const float* __restrict__ cp,
                                                 float* __restrict__ out, int nblk) {
    __shared__ float sl[256], sc[256];
    int tid = threadIdx.x;
    float a = 0.0f, c = 0.0f;
    for (int i = tid; i < nblk; i += 256) { a += lp[i]; c += cp[i]; }
    sl[tid] = a; sc[tid] = c;
    __syncthreads();
    for (int st = 128; st > 0; st >>= 1) {
        if (tid < st) { sl[tid] += sl[tid + st]; sc[tid] += sc[tid + st]; }
        __syncthreads();
    }
    if (tid == 0) {
        const float invN = 1.0f / (float)((size_t)BATCH * NQ);
        out[(size_t)BATCH * NQ]     = sl[0] * invN;  // loss
        out[(size_t)BATCH * NQ + 1] = sc[0] * invN;  // accuracy
    }
}

extern "C" void kernel_launch(void* const* d_in, const int* in_sizes, int n_in,
                              void* d_out, int out_size, void* d_ws, size_t ws_size,
                              hipStream_t stream) {
    const float* sup  = (const float*)d_in[0];   // (B, NSUP, D) f32
    const float* qry  = (const float*)d_in[1];   // (B, NQ, D)  f32
    const int*   stgt = (const int*)d_in[2];     // (B, NSUP)
    const int*   qtgt = (const int*)d_in[3];     // (B, NQ)
    float* out = (float*)d_out;                  // preds (B*NQ) + loss + acc

    char* ws = (char*)d_ws;
    const size_t SUMS_F = (size_t)BATCH * NCLS * DIM;     // 1,048,576 floats
    const size_t CNT_F  = (size_t)BATCH * NCLS;           // 2048
    float* sums   = (float*)(ws);
    float* counts = (float*)(ws + SUMS_F * 4);
    float* p_sq   = (float*)(ws + (SUMS_F + CNT_F) * 4);
    float* q_sq   = (float*)(ws + (SUMS_F + 2 * CNT_F) * 4);
    float* loss_p = (float*)(ws + (SUMS_F + 2 * CNT_F + (size_t)BATCH * NQ) * 4);
    float* corr_p = loss_p + 1024;
    unsigned short* protoBf = (unsigned short*)(corr_p + 1024);
    unsigned short* qBf     = protoBf + SUMS_F;           // 16B-aligned (offset 6,578,176)

    const int nZero = (int)(SUMS_F + CNT_F);
    pn_zero<<<(nZero + 255) / 256, 256, 0, stream>>>(sums, nZero);
    pn_scatter<<<BATCH * NSUP, 256, 0, stream>>>(sup, stgt, sums, counts);
    pn_finalize<<<BATCH * NCLS, 256, 0, stream>>>(sums, counts, protoBf, p_sq);
    pn_qprep<<<BATCH * NQ, 256, 0, stream>>>(qry, q_sq, qBf);
    const int nGemmBlk = BATCH * (NQ / TILE_Q);           // 1024
    pn_gemm<<<nGemmBlk, 256, 0, stream>>>(qBf, qtgt, protoBf, p_sq, q_sq,
                                          out, loss_p, corr_p);
    pn_reduce<<<1, 256, 0, stream>>>(loss_p, corr_p, out, nGemmBlk);
}
